// NeRF_52621939310733
// MI455X (gfx1250) — matrix operand order repaired and checked by reading, not compiled, not run on hardware
//
#include <hip/hip_runtime.h>
#include <math.h>
#include <stdint.h>

// ---------------------------------------------------------------------------
// Types for CDNA5 WMMA / TDM
// ---------------------------------------------------------------------------
typedef __attribute__((ext_vector_type(8)))  _Float16 v8h;
typedef __attribute__((ext_vector_type(16))) _Float16 v16h;
typedef __attribute__((ext_vector_type(8)))  float    v8f;
typedef __attribute__((ext_vector_type(4)))  unsigned int u32x4;
typedef __attribute__((ext_vector_type(8)))  int      i32x8;
typedef __attribute__((ext_vector_type(4)))  int      i32x4;

#if defined(__AMDGCN__) && __has_builtin(__builtin_amdgcn_tensor_load_to_lds) && \
    __has_builtin(__builtin_amdgcn_s_wait_tensorcnt)
#define USE_TDM 1
#else
#define USE_TDM 0
#endif

// ---------------------------------------------------------------------------
// Problem constants (match reference)
// ---------------------------------------------------------------------------
constexpr int   BRAYS = 1024;
constexpr int   NC    = 64;           // coarse samples / ray
constexpr int   NFT   = 192;          // fine samples / ray (N_F + N_C)
constexpr int   PTS_C = BRAYS * NC;   // 65536
constexpr int   PTS_F = BRAYS * NFT;  // 196608
constexpr float T_N = 0.0f, T_F = 2.5f;

// LDS activation row stride (f16 elems): max K = 256 + 64 (gx concat)
constexpr int LDA = 320;
// panel: [N][32] f16, double buffered, N<=256
constexpr int PANEL_MAX = 256 * 32;

// Transposed/padded f16 weight block offsets (f16 elements), layout [N][Kpad]
constexpr int OFF_W0    = 0;        // 256 x 64   (K 60 -> 64)
constexpr int OFF_W1    = 16384;    // 256 x 256
constexpr int OFF_W2    = 81920;
constexpr int OFF_W3    = 147456;
constexpr int OFF_W4    = 212992;
constexpr int OFF_W5    = 278528;   // 256 x 320  (K 316 -> 320)
constexpr int OFF_W6    = 360448;
constexpr int OFF_W7    = 425984;
constexpr int OFF_WFEAT = 491520;   // 256 x 256
constexpr int OFF_WSIG  = 557056;   // 16  x 256  (N 1 -> 16)
constexpr int OFF_WRGB1 = 561152;   // 128 x 288  (K 280 -> 288)
constexpr int OFF_WRGB2 = 598016;   // 16  x 128  (N 3 -> 16)
constexpr int WT_TOTAL  = 600064;

// Padded f32 bias offsets (f32 elements)
constexpr int BOFF_B0 = 0,  BOFF_B1 = 256, BOFF_B2 = 512,  BOFF_B3 = 768;
constexpr int BOFF_B4 = 1024, BOFF_B5 = 1280, BOFF_B6 = 1536, BOFF_B7 = 1792;
constexpr int BOFF_BFEAT = 2048, BOFF_BSIG = 2304, BOFF_BRGB1 = 2320, BOFF_BRGB2 = 2448;
constexpr int B_TOTAL = 2464;

// ---------------------------------------------------------------------------
// Hash RNG (stand-in for jax threefry; deterministic)
// ---------------------------------------------------------------------------
__device__ __forceinline__ float u01(unsigned x) {
  x ^= x >> 17; x *= 0xed5ad4bbu;
  x ^= x >> 11; x *= 0xac4c1b51u;
  x ^= x >> 15; x *= 0x31848babu;
  x ^= x >> 14;
  return (x >> 8) * (1.0f / 16777216.0f);
}

// ---------------------------------------------------------------------------
// WMMA fragment loaders (contiguous k = 16*(lane/16) + s in both A and B, wave32)
// ---------------------------------------------------------------------------
__device__ __forceinline__ v16h load_a_frag(const _Float16* act, int mbase,
                                            int kbase, int lane) {
  int row  = mbase + (lane & 15);
  int half = lane >> 4;
  const _Float16* p = act + row * LDA + kbase + half * 16;
  v8h lo = *(const v8h*)p;
  v8h hi = *(const v8h*)(p + 8);
  v16h a;
#pragma unroll
  for (int i = 0; i < 8; i++) { a[i] = lo[i]; a[8 + i] = hi[i]; }
  return a;
}

// B fragment from global transposed weights [N][Kpad]
template <int KP>
__device__ __forceinline__ v16h load_b_frag(const _Float16* __restrict__ Wt,
                                            int nbase, int kbase, int lane) {
  int n  = nbase + (lane & 15);
  int ks = kbase + (lane >> 4) * 16;
  return *(const v16h*)(Wt + (size_t)n * KP + ks);
}

// B fragment from LDS panel [N][32]
__device__ __forceinline__ v16h load_panel_frag(const _Float16* panel,
                                                int nbase, int lane) {
  int n  = nbase + (lane & 15);
  int ks = (lane >> 4) * 16;
  const _Float16* p = panel + n * 32 + ks;
  v8h lo = *(const v8h*)p;
  v8h hi = *(const v8h*)(p + 8);
  v16h b;
#pragma unroll
  for (int i = 0; i < 8; i++) { b[i] = lo[i]; b[8 + i] = hi[i]; }
  return b;
}

// ---------------------------------------------------------------------------
// TDM: stage one [N][32]-f16 weight panel (k-slab) into LDS.
// D# per CDNA5 ISA (2D tile; groups 2/3 zero).
// ---------------------------------------------------------------------------
#if USE_TDM
template <int Kpad, int N>
__device__ __forceinline__ void tdm_issue(const _Float16* Wt, int slab,
                                          unsigned lds_byte) {
  unsigned long long ga = (unsigned long long)(uintptr_t)(Wt + slab * 32);
  u32x4 g0;
  g0[0] = 1u;                                      // count=1, user mode
  g0[1] = lds_byte;                                // lds_addr (bytes)
  g0[2] = (unsigned)ga;                            // global_addr[31:0]
  g0[3] = ((unsigned)(ga >> 32) & 0x01FFFFFFu) | (2u << 30);  // addr[56:32]|type=2
  i32x8 g1;
  g1[0] = 0x00010000;                              // wg_mask=0, data_size=2B
  g1[1] = (int)((unsigned)(Kpad & 0xFFFF) << 16);  // tensor_dim0[15:0]
  g1[2] = (int)(((unsigned)Kpad >> 16) | ((unsigned)(N & 0xFFFF) << 16)); // td0 hi | td1 lo
  g1[3] = (int)(((unsigned)N >> 16) | (32u << 16));// td1 hi | tile_dim0=32
  g1[4] = (int)(N & 0xFFFF);                       // tile_dim1=N, tile_dim2=0
  g1[5] = Kpad;                                    // tensor_dim0_stride lo
  g1[6] = 0;
  g1[7] = 0;
  i32x4 z4 = {0, 0, 0, 0};
#if __clang_major__ >= 23
  i32x8 z8 = {0, 0, 0, 0, 0, 0, 0, 0};
  __builtin_amdgcn_tensor_load_to_lds(g0, g1, z4, z4, z8, 0);
#else
  __builtin_amdgcn_tensor_load_to_lds(g0, g1, z4, z4, 0);
#endif
}
#endif

// ---------------------------------------------------------------------------
// Dense layer: act[64 x K] (LDS f16) x Wt[N x K] -> act[64 x N] in place.
// k-outer / n-tile-inner; each wave: fixed M-tile, TPW N-tiles, A loaded once
// per k-slab. Weight panels staged by TDM (double buffered) when available.
// ---------------------------------------------------------------------------
template <int K, int N, int RELU>
__device__ __forceinline__ void dense_layer(_Float16* act, _Float16* panel,
                                            const _Float16* __restrict__ Wt,
                                            const float* __restrict__ bias) {
  static_assert((K % 32) == 0 && (N % 16) == 0 && N >= 32, "shape");
  constexpr int NK  = K / 32;
  constexpr int TPW = (4 * (N / 16)) / 8;   // n-tiles per wave (8 waves)
  constexpr int PANEL = N * 32;             // f16 elems per panel buffer
  int lane  = threadIdx.x & 31;
  int wave  = threadIdx.x >> 5;
  int mtile = wave & 3;
  int npar  = wave >> 2;                    // n-tile parity
  v8f acc[TPW];
  v8f zero = {};
#pragma unroll
  for (int i = 0; i < TPW; i++) acc[i] = zero;

#if USE_TDM
  if (wave == 0)
    tdm_issue<K, N>(Wt, 0, (unsigned)(uintptr_t)panel);
  for (int j = 0; j < NK; j++) {
    if (wave == 0) __builtin_amdgcn_s_wait_tensorcnt(0);
    __syncthreads();                         // panel j staged; old buffer free
    if (wave == 0 && j + 1 < NK)
      tdm_issue<K, N>(Wt, j + 1,
                      (unsigned)(uintptr_t)(panel + ((j + 1) & 1) * PANEL));
    const _Float16* pb = panel + (j & 1) * PANEL;
    v16h a = load_a_frag(act, mtile * 16, j * 32, lane);
#pragma unroll
    for (int i = 0; i < TPW; i++) {
      int ntile = npar + i * 2;
      v16h b = load_panel_frag(pb, ntile * 16, lane);
      acc[i] = __builtin_amdgcn_wmma_f32_16x16x32_f16(
          false, a, false, b, (short)0, acc[i], false, false);
    }
  }
#else
  (void)panel;
#pragma unroll
  for (int j = 0; j < NK; j++) {
    v16h a = load_a_frag(act, mtile * 16, j * 32, lane);
#pragma unroll
    for (int i = 0; i < TPW; i++) {
      int ntile = npar + i * 2;
      v16h b = load_b_frag<K>(Wt, ntile * 16, j * 32, lane);
      acc[i] = __builtin_amdgcn_wmma_f32_16x16x32_f16(
          false, a, false, b, (short)0, acc[i], false, false);
    }
  }
#endif

  // bias + activation in registers
#pragma unroll
  for (int i = 0; i < TPW; i++) {
    int n = (npar + i * 2) * 16 + (lane & 15);
    float bn = bias[n];
#pragma unroll
    for (int r = 0; r < 8; r++) {
      float v = acc[i][r] + bn;
      if (RELU) v = v > 0.f ? v : 0.f;
      acc[i][r] = v;
    }
  }
  __syncthreads();   // all reads of act complete
#pragma unroll
  for (int i = 0; i < TPW; i++) {
    int n  = (npar + i * 2) * 16 + (lane & 15);
    int mb = mtile * 16 + ((lane >> 4) << 3);
#pragma unroll
    for (int r = 0; r < 8; r++)
      act[(mb + r) * LDA + n] = (_Float16)acc[i][r];
  }
  __syncthreads();   // writes visible before next layer reads
}

// ---------------------------------------------------------------------------
// N=16 head layers: OP==0 -> sigma = relu(col 0) to out[p];
//                   OP==1 -> rgb   = sigmoid(cols 0..2) to out[p*3+n]
// ---------------------------------------------------------------------------
template <int K, int OP>
__device__ __forceinline__ void head16(const _Float16* act,
                                       const _Float16* __restrict__ Wt,
                                       const float* __restrict__ bias,
                                       float* __restrict__ out, int pointBase) {
  int lane = threadIdx.x & 31;
  int wave = threadIdx.x >> 5;
  if (wave < 4) {
    int mtile = wave;
    v8f acc = {};
#pragma unroll
    for (int k = 0; k < K; k += 32) {
      v16h a = load_a_frag(act, mtile * 16, k, lane);
      v16h b = load_b_frag<K>(Wt, 0, k, lane);
      acc = __builtin_amdgcn_wmma_f32_16x16x32_f16(
          false, a, false, b, (short)0, acc, false, false);
    }
    int n  = lane & 15;
    float bn = bias[n];
    int mb = mtile * 16 + ((lane >> 4) << 3);
#pragma unroll
    for (int r = 0; r < 8; r++) {
      float v = acc[r] + bn;
      int p = pointBase + mb + r;
      if (OP == 0) {
        if (n == 0) out[p] = v > 0.f ? v : 0.f;
      } else {
        if (n < 3) out[(size_t)p * 3 + n] = 1.f / (1.f + __expf(-v));
      }
    }
  }
  __syncthreads();
}

// ---------------------------------------------------------------------------
// Main WMMA MLP: one block = 64 points through the whole radiance field.
// ---------------------------------------------------------------------------
__global__ __launch_bounds__(256) void NeRF_mlp_wmma(
    const _Float16* __restrict__ WT, const float* __restrict__ BIAS,
    const _Float16* __restrict__ gx, const _Float16* __restrict__ gd,
    float* __restrict__ sigma_out, float* __restrict__ rgb_out, int npts) {
  alignas(32) __shared__ _Float16 sAct[64 * LDA];        // 40 KB
  alignas(32) __shared__ _Float16 sPanel[2 * PANEL_MAX]; // 32 KB
  int tid = threadIdx.x;
  int pointBase = blockIdx.x * 64;
  if (pointBase >= npts) return;

  __builtin_prefetch(WT, 0, 1);   // warm path (global_prefetch_b8)

  // stage gx (padded 64) into A cols 0..63
  for (int i = tid; i < 64 * 64; i += 256) {
    int m = i >> 6, c = i & 63;
    sAct[m * LDA + c] = gx[(size_t)(pointBase + m) * 64 + c];
  }
  __syncthreads();

  dense_layer<64, 256, 1>(sAct, sPanel, WT + OFF_W0, BIAS + BOFF_B0);
  dense_layer<256, 256, 1>(sAct, sPanel, WT + OFF_W1, BIAS + BOFF_B1);
  dense_layer<256, 256, 1>(sAct, sPanel, WT + OFF_W2, BIAS + BOFF_B2);
  dense_layer<256, 256, 1>(sAct, sPanel, WT + OFF_W3, BIAS + BOFF_B3);
  dense_layer<256, 256, 1>(sAct, sPanel, WT + OFF_W4, BIAS + BOFF_B4);

  // skip-concat: [h, gx] -> cols 256..319
  for (int i = tid; i < 64 * 64; i += 256) {
    int m = i >> 6, c = i & 63;
    sAct[m * LDA + 256 + c] = gx[(size_t)(pointBase + m) * 64 + c];
  }
  __syncthreads();

  dense_layer<320, 256, 1>(sAct, sPanel, WT + OFF_W5, BIAS + BOFF_B5);
  dense_layer<256, 256, 1>(sAct, sPanel, WT + OFF_W6, BIAS + BOFF_B6);
  dense_layer<256, 256, 1>(sAct, sPanel, WT + OFF_W7, BIAS + BOFF_B7);

  // sigma head (reads h; no LDS writes)
  head16<256, 0>(sAct, WT + OFF_WSIG, BIAS + BOFF_BSIG, sigma_out, pointBase);

  // feat = h @ W_feat + b (no relu), in place
  dense_layer<256, 256, 0>(sAct, sPanel, WT + OFF_WFEAT, BIAS + BOFF_BFEAT);

  // concat gd (padded 32) -> cols 256..287
  for (int i = tid; i < 64 * 32; i += 256) {
    int m = i >> 5, c = i & 31;
    sAct[m * LDA + 256 + c] = gd[(size_t)(pointBase + m) * 32 + c];
  }
  __syncthreads();

  dense_layer<288, 128, 1>(sAct, sPanel, WT + OFF_WRGB1, BIAS + BOFF_BRGB1);
  head16<128, 1>(sAct, WT + OFF_WRGB2, BIAS + BOFF_BRGB2, rgb_out, pointBase);
}

// ---------------------------------------------------------------------------
// Weight transpose/pad f32 -> f16  (dst layout [Npad][Kpad])
// ---------------------------------------------------------------------------
__global__ void NeRF_conv_w(const float* __restrict__ src, _Float16* __restrict__ dst,
                            int K, int Nsrc, int Npad, int Kpad) {
  int i = blockIdx.x * blockDim.x + threadIdx.x;
  int total = Npad * Kpad;
  if (i >= total) return;
  int n = i / Kpad, k = i - n * Kpad;
  float v = (k < K && n < Nsrc) ? src[(size_t)k * Nsrc + n] : 0.f;
  dst[i] = (_Float16)v;
}

__global__ void NeRF_conv_b(const float* __restrict__ src, float* __restrict__ dst,
                            int N, int Npad) {
  int i = blockIdx.x * blockDim.x + threadIdx.x;
  if (i >= Npad) return;
  dst[i] = (i < N) ? src[i] : 0.f;
}

// ---------------------------------------------------------------------------
// Stratified coarse sampling
// ---------------------------------------------------------------------------
__global__ void NeRF_coarse_sample(float* __restrict__ t_c) {
  int r = blockIdx.x * blockDim.x + threadIdx.x;
  if (r >= BRAYS) return;
  for (int i = 0; i < NC; i++) {
    float lo = T_N + (T_F - T_N) * (float)i / (float)NC;
    float hi = T_N + (T_F - T_N) * (float)(i + 1) / (float)NC;
    float u = u01(0x9E3779B9u + (unsigned)(r * NC + i));
    t_c[r * NC + i] = lo + (hi - lo) * u;
  }
}

// ---------------------------------------------------------------------------
// Positional encoding: gx (L=10 -> 60, pad 64), gd (L=4 -> 24, pad 32)
// ---------------------------------------------------------------------------
__global__ void NeRF_encode(const float* __restrict__ o, const float* __restrict__ dvec,
                            const float* __restrict__ t, _Float16* __restrict__ gx,
                            _Float16* __restrict__ gd, int Ns, int npts) {
  int p = blockIdx.x * blockDim.x + threadIdx.x;
  if (p >= npts) return;
  int r = p / Ns;
  float tv = t[p];
  float x0 = o[r * 3 + 0] + tv * dvec[r * 3 + 0];
  float x1 = o[r * 3 + 1] + tv * dvec[r * 3 + 1];
  float x2 = o[r * 3 + 2] + tv * dvec[r * 3 + 2];
  _Float16* gxp = gx + (size_t)p * 64;
  float f = 3.14159265358979323846f;
#pragma unroll
  for (int l = 0; l < 10; l++) {
    gxp[l * 6 + 0] = (_Float16)__sinf(x0 * f);
    gxp[l * 6 + 1] = (_Float16)__sinf(x1 * f);
    gxp[l * 6 + 2] = (_Float16)__sinf(x2 * f);
    gxp[l * 6 + 3] = (_Float16)__cosf(x0 * f);
    gxp[l * 6 + 4] = (_Float16)__cosf(x1 * f);
    gxp[l * 6 + 5] = (_Float16)__cosf(x2 * f);
    f *= 2.f;
  }
  for (int i = 60; i < 64; i++) gxp[i] = (_Float16)0.f;

  float d0 = dvec[r * 3 + 0], d1 = dvec[r * 3 + 1], d2 = dvec[r * 3 + 2];
  _Float16* gdp = gd + (size_t)p * 32;
  f = 3.14159265358979323846f;
#pragma unroll
  for (int l = 0; l < 4; l++) {
    gdp[l * 6 + 0] = (_Float16)__sinf(d0 * f);
    gdp[l * 6 + 1] = (_Float16)__sinf(d1 * f);
    gdp[l * 6 + 2] = (_Float16)__sinf(d2 * f);
    gdp[l * 6 + 3] = (_Float16)__cosf(d0 * f);
    gdp[l * 6 + 4] = (_Float16)__cosf(d1 * f);
    gdp[l * 6 + 5] = (_Float16)__cosf(d2 * f);
    f *= 2.f;
  }
  for (int i = 24; i < 32; i++) gdp[i] = (_Float16)0.f;
}

// ---------------------------------------------------------------------------
// Volume rendering composite (one thread per ray)
// ---------------------------------------------------------------------------
__global__ void NeRF_composite(const float* __restrict__ t, const float* __restrict__ sigma,
                               const float* __restrict__ rgb, float* __restrict__ w_out,
                               float* __restrict__ C_out, int Ns) {
  int r = blockIdx.x * blockDim.x + threadIdx.x;
  if (r >= BRAYS) return;
  float T = 1.f, c0 = 0.f, c1 = 0.f, c2 = 0.f, wsum = 0.f;
  for (int i = 0; i < Ns; i++) {
    int idx = r * Ns + i;
    float delta = (i + 1 < Ns) ? (t[idx + 1] - t[idx]) : 1.0e8f;
    float mass = sigma[idx] * delta;
    float e = __expf(-mass);
    float w = T * (1.f - e);
    T *= e;
    c0 += w * rgb[(size_t)idx * 3 + 0];
    c1 += w * rgb[(size_t)idx * 3 + 1];
    c2 += w * rgb[(size_t)idx * 3 + 2];
    wsum += w;
    if (w_out) w_out[idx] = w;
  }
  float bg = 1.f - wsum;   // BG = 1
  C_out[r * 3 + 0] = c0 + bg;
  C_out[r * 3 + 1] = c1 + bg;
  C_out[r * 3 + 2] = c2 + bg;
}

// ---------------------------------------------------------------------------
// Inverse-CDF fine sampling + merge/sort with t_c (one thread per ray)
// ---------------------------------------------------------------------------
__global__ void NeRF_fine_sample(const float* __restrict__ t_c,
                                 const float* __restrict__ w_c,
                                 float* __restrict__ t_f) {
  int r = blockIdx.x * blockDim.x + threadIdx.x;
  if (r >= BRAYS) return;
  float cdf[NC + 1];
  cdf[0] = 0.f;
  float acc = 0.f;
  for (int i = 0; i < NC; i++) {
    float w = w_c[r * NC + i];
    if (w < 1e-16f) w = 1e-16f;
    acc += w;
    cdf[i + 1] = acc;
  }
  float inv = 1.f / acc;
  for (int i = 1; i <= NC; i++) cdf[i] *= inv;

  float tf[NFT];
  for (int i = 0; i < NC; i++) tf[i] = t_c[r * NC + i];
  for (int j = 0; j < NFT - NC; j++) {
    float u = u01(0x85EBCA6Bu + (unsigned)(r * 128 + j));
    int lo = 0, hi = NC + 1;
    while (lo < hi) {
      int mid = (lo + hi) >> 1;
      if (u < cdf[mid]) hi = mid; else lo = mid + 1;
    }
    int idx = lo;
    if (idx < 1) idx = 1;
    if (idx > NC) idx = NC;
    float clo = cdf[idx - 1], chi = cdf[idx];
    float plo = T_N + (T_F - T_N) * (float)(idx - 1) / (float)NC;
    float phi = T_N + (T_F - T_N) * (float)idx / (float)NC;
    float den = (chi - clo < 1e-10f) ? 1.f : (chi - clo);
    tf[NC + j] = plo + (u - clo) / den * (phi - plo);
  }
  for (int i = 1; i < NFT; i++) {
    float v = tf[i];
    int j = i - 1;
    while (j >= 0 && tf[j] > v) { tf[j + 1] = tf[j]; j--; }
    tf[j + 1] = v;
  }
  for (int i = 0; i < NFT; i++) t_f[r * NFT + i] = tf[i];
}

// ---------------------------------------------------------------------------
// Host: kernel_launch
// ---------------------------------------------------------------------------
struct MD { int rel; int K; int Nsrc; int Npad; int Kpad; int dstOff; };
struct BD { int rel; int N; int Npad; int dstOff; };

extern "C" void kernel_launch(void* const* d_in, const int* in_sizes, int n_in,
                              void* d_out, int out_size, void* d_ws, size_t ws_size,
                              hipStream_t stream) {
  (void)in_sizes; (void)n_in; (void)out_size; (void)ws_size;
  const float* o    = (const float*)d_in[0];
  const float* dvec = (const float*)d_in[1];
  const int setBase[2] = {2, 26};

  static const MD mats[12] = {
      { 0,  60, 256, 256,  64, OFF_W0   },
      { 2, 256, 256, 256, 256, OFF_W1   },
      { 4, 256, 256, 256, 256, OFF_W2   },
      { 6, 256, 256, 256, 256, OFF_W3   },
      { 8, 256, 256, 256, 256, OFF_W4   },
      {10, 316, 256, 256, 320, OFF_W5   },
      {12, 256, 256, 256, 256, OFF_W6   },
      {14, 256, 256, 256, 256, OFF_W7   },
      {16, 256, 256, 256, 256, OFF_WFEAT},
      {18, 280, 128, 128, 288, OFF_WRGB1},
      {20, 128,   3,  16, 128, OFF_WRGB2},
      {22, 256,   1,  16, 256, OFF_WSIG },
  };
  static const BD bds[12] = {
      { 1, 256, 256, BOFF_B0  }, { 3, 256, 256, BOFF_B1  },
      { 5, 256, 256, BOFF_B2  }, { 7, 256, 256, BOFF_B3  },
      { 9, 256, 256, BOFF_B4  }, {11, 256, 256, BOFF_B5  },
      {13, 256, 256, BOFF_B6  }, {15, 256, 256, BOFF_B7  },
      {17, 256, 256, BOFF_BFEAT}, {19, 128, 128, BOFF_BRGB1},
      {21,   3,  16, BOFF_BRGB2}, {23,   1,  16, BOFF_BSIG },
  };

  // Workspace carve (256B aligned)
  char* wsp = (char*)d_ws;
  auto alloc = [&](size_t bytes) -> void* {
    void* p = wsp;
    wsp += (bytes + 255) & ~(size_t)255;
    return p;
  };
  _Float16* WT[2];  float* BI[2];
  WT[0] = (_Float16*)alloc((size_t)WT_TOTAL * 2);
  WT[1] = (_Float16*)alloc((size_t)WT_TOTAL * 2);
  BI[0] = (float*)alloc((size_t)B_TOTAL * 4);
  BI[1] = (float*)alloc((size_t)B_TOTAL * 4);
  float*    t_c   = (float*)alloc((size_t)PTS_C * 4);
  float*    w_c   = (float*)alloc((size_t)PTS_C * 4);
  float*    t_f   = (float*)alloc((size_t)PTS_F * 4);
  _Float16* gx_c  = (_Float16*)alloc((size_t)PTS_C * 64 * 2);
  _Float16* gd_c  = (_Float16*)alloc((size_t)PTS_C * 32 * 2);
  _Float16* gx_f  = (_Float16*)alloc((size_t)PTS_F * 64 * 2);
  _Float16* gd_f  = (_Float16*)alloc((size_t)PTS_F * 32 * 2);
  float*    sig_c = (float*)alloc((size_t)PTS_C * 4);
  float*    rgb_c = (float*)alloc((size_t)PTS_C * 3 * 4);
  float*    sig_f = (float*)alloc((size_t)PTS_F * 4);
  float*    rgb_f = (float*)alloc((size_t)PTS_F * 3 * 4);

  // 1) convert weights/biases for both parameter sets
  for (int s = 0; s < 2; s++) {
    for (int m = 0; m < 12; m++) {
      const MD& md = mats[m];
      const float* src = (const float*)d_in[setBase[s] + md.rel];
      int total = md.Npad * md.Kpad;
      NeRF_conv_w<<<(total + 255) / 256, 256, 0, stream>>>(
          src, WT[s] + md.dstOff, md.K, md.Nsrc, md.Npad, md.Kpad);
    }
    for (int m = 0; m < 12; m++) {
      const BD& bd = bds[m];
      const float* src = (const float*)d_in[setBase[s] + bd.rel];
      NeRF_conv_b<<<(bd.Npad + 255) / 256, 256, 0, stream>>>(
          src, BI[s] + bd.dstOff, bd.N, bd.Npad);
    }
  }

  float* C_c = (float*)d_out;            // 1024 x 3
  float* C_f = (float*)d_out + BRAYS * 3;

  // 2) coarse pass
  NeRF_coarse_sample<<<(BRAYS + 255) / 256, 256, 0, stream>>>(t_c);
  NeRF_encode<<<(PTS_C + 255) / 256, 256, 0, stream>>>(o, dvec, t_c, gx_c, gd_c, NC, PTS_C);
  NeRF_mlp_wmma<<<PTS_C / 64, 256, 0, stream>>>(WT[0], BI[0], gx_c, gd_c, sig_c, rgb_c, PTS_C);
  NeRF_composite<<<(BRAYS + 255) / 256, 256, 0, stream>>>(t_c, sig_c, rgb_c, w_c, C_c, NC);

  // 3) fine pass
  NeRF_fine_sample<<<(BRAYS + 255) / 256, 256, 0, stream>>>(t_c, w_c, t_f);
  NeRF_encode<<<(PTS_F + 255) / 256, 256, 0, stream>>>(o, dvec, t_f, gx_f, gd_f, NFT, PTS_F);
  NeRF_mlp_wmma<<<PTS_F / 64, 256, 0, stream>>>(WT[1], BI[1], gx_f, gd_f, sig_f, rgb_f, PTS_F);
  NeRF_composite<<<(BRAYS + 255) / 256, 256, 0, stream>>>(t_f, sig_f, rgb_f, (float*)nullptr, C_f, NFT);
}
